// RealNVP_65841848648440
// MI455X (gfx1250) — compile-verified, matrix operand order checked by reading, not run on hardware
//
#include <hip/hip_runtime.h>

// ---------------- problem constants ----------------
#define B_N 131072
#define D_N 64
#define H_N 256
#define L_N 8

// LDS activation row stride (bf16 elems), padded to spread banks
#define STR 264

// packed bf16 weight layout (elements), per layer
#define OFF_SW1 0        // W1: N=256, K=64  -> 32 frags  * 512 = 16384
#define OFF_SW2 16384    // W2: N=256, K=256 -> 128 frags * 512 = 65536
#define OFF_SW3 81920    // W3: N=64,  K=256 -> 32 frags  * 512 = 16384
#define OFF_TW1 98304
#define OFF_TW2 114688
#define OFF_TW3 180224
#define PL      196608   // per-layer packed elements

typedef __attribute__((ext_vector_type(16))) __bf16 bf16x16;
typedef __attribute__((ext_vector_type(8)))  __bf16 bf16x8;
typedef __attribute__((ext_vector_type(8)))  float  f32x8;

// Native CDNA5 tanh (TRANS pipe) — builtin if declared, else raw ISA with
// the required post-TRANS spacing (1 indep op before result use).
#if __has_builtin(__builtin_amdgcn_tanhf)
__device__ __forceinline__ float fast_tanh(float x) {
  return __builtin_amdgcn_tanhf(x);
}
#else
__device__ __forceinline__ float fast_tanh(float x) {
  float r;
  asm volatile("v_tanh_f32 %0, %1\n\tv_nop\n\tv_nop" : "=v"(r) : "v"(x));
  return r;
}
#endif

// ---- A fragment (16x32 bf16) from row-major LDS tile -----------------------
// ISA 16-bit A layout: lane L(0-15): row M=L, slots 0-7 -> K=k0..k0+7,
// slots 8-15 -> K=k0+16..k0+23 with k0 = 8*(lane>>4) + 32*kt.
__device__ __forceinline__ bf16x16 load_a_frag(const __bf16* buf,
                                               int kt, int g, int ln) {
  const int k0 = kt * 32 + 8 * g;
  const __bf16* p = buf + ln * STR + k0;
  bf16x8 lo = *(const bf16x8*)(p);
  bf16x8 hi = *(const bf16x8*)(p + 16);
  return __builtin_shufflevector(lo, hi, 0,1,2,3,4,5,6,7,8,9,10,11,12,13,14,15);
}

// ---- B fragment (32x16 bf16) from packed weights: lane-major, 16 contiguous
__device__ __forceinline__ bf16x16 load_b_frag(const __bf16* w,
                                               int nt, int kt, int nkt, int lane) {
  const __bf16* p = w + (((size_t)(nt * nkt + kt)) << 9) + lane * 16;
  bf16x8 lo = *(const bf16x8*)(p);
  bf16x8 hi = *(const bf16x8*)(p + 8);
  return __builtin_shufflevector(lo, hi, 0,1,2,3,4,5,6,7,8,9,10,11,12,13,14,15);
}

// ---- one MLP stage: out = tanh(in @ W^T + bias [+ in residual]) -> LDS ----
template <int NNT, int NKT, bool RESID>
__device__ __forceinline__ void mlp_stage(const __bf16* inB, __bf16* outB,
                                          const __bf16* wmat, const float* bias,
                                          int lane, int g, int ln) {
  bf16x16 af[NKT];
#pragma unroll
  for (int kt = 0; kt < NKT; ++kt) af[kt] = load_a_frag(inB, kt, g, ln);
#pragma unroll
  for (int nt = 0; nt < NNT; ++nt) {
    float bb = bias[nt * 16 + ln];
    f32x8 acc;
#pragma unroll
    for (int j = 0; j < 8; ++j) acc[j] = bb;
    if (RESID) {
#pragma unroll
      for (int j = 0; j < 8; ++j)
        acc[j] += (float)inB[(j + 8 * g) * STR + nt * 16 + ln];
    }
#pragma unroll
    for (int kt = 0; kt < NKT; ++kt) {
      bf16x16 bm = load_b_frag(wmat, nt, kt, NKT, lane);
      acc = __builtin_amdgcn_wmma_f32_16x16x32_bf16(
          false, af[kt], false, bm, (short)0, acc, false, false);
    }
#pragma unroll
    for (int j = 0; j < 8; ++j)
      outB[(j + 8 * g) * STR + nt * 16 + ln] = (__bf16)fast_tanh(acc[j]);
  }
  asm volatile("s_wait_dscnt 0" ::: "memory");   // close same-wave LDS RAW
}

// ---- final projection stage -> f32 registers (no activation) --------------
template <int NKT>
__device__ __forceinline__ void mlp_stage_out(const __bf16* inB,
                                              const __bf16* wmat, const float* bias,
                                              float res[4][8],
                                              int lane, int g, int ln) {
  bf16x16 af[NKT];
#pragma unroll
  for (int kt = 0; kt < NKT; ++kt) af[kt] = load_a_frag(inB, kt, g, ln);
#pragma unroll
  for (int nt = 0; nt < 4; ++nt) {
    float bb = bias[nt * 16 + ln];
    f32x8 acc;
#pragma unroll
    for (int j = 0; j < 8; ++j) acc[j] = bb;
#pragma unroll
    for (int kt = 0; kt < NKT; ++kt) {
      bf16x16 bm = load_b_frag(wmat, nt, kt, NKT, lane);
      acc = __builtin_amdgcn_wmma_f32_16x16x32_bf16(
          false, af[kt], false, bm, (short)0, acc, false, false);
    }
#pragma unroll
    for (int j = 0; j < 8; ++j) res[nt][j] = acc[j];
  }
  asm volatile("s_wait_dscnt 0" ::: "memory");
}

// ---------------- weight packing: f32 [N,K] -> bf16 B-fragments ------------
__global__ __launch_bounds__(256)
void pack_weights_kernel(const float* __restrict__ sW1, const float* __restrict__ sW2,
                         const float* __restrict__ sW3, const float* __restrict__ tW1,
                         const float* __restrict__ tW2, const float* __restrict__ tW3,
                         __bf16* __restrict__ wp) {
  int tid = blockIdx.x * 256 + threadIdx.x;
  if (tid >= L_N * PL) return;
  int l = tid / PL;
  int r = tid % PL;
  const float* W; int Kdim, off;
  if      (r < OFF_SW2) { W = sW1 + l * H_N * D_N; Kdim = D_N; off = OFF_SW1; }
  else if (r < OFF_SW3) { W = sW2 + l * H_N * H_N; Kdim = H_N; off = OFF_SW2; }
  else if (r < OFF_TW1) { W = sW3 + l * D_N * H_N; Kdim = H_N; off = OFF_SW3; }
  else if (r < OFF_TW2) { W = tW1 + l * H_N * D_N; Kdim = D_N; off = OFF_TW1; }
  else if (r < OFF_TW3) { W = tW2 + l * H_N * H_N; Kdim = H_N; off = OFF_TW2; }
  else                  { W = tW3 + l * D_N * H_N; Kdim = H_N; off = OFF_TW3; }
  int e = r - off;
  int nkt  = Kdim >> 5;
  int frag = e >> 9;
  int lane = (e >> 4) & 31;
  int s    = e & 15;
  int nt = frag / nkt;
  int kt = frag - nt * nkt;
  int n = nt * 16 + (lane & 15);
  int k = ((s < 8) ? s : s + 8) + 8 * (lane >> 4) + kt * 32;
  wp[tid] = (__bf16)W[n * Kdim + k];   // B[k][n] = W[n][k]  (W used as A@W^T)
}

// ---------------- main flow kernel: 2 waves/block, 16 rows/wave ------------
__global__ __launch_bounds__(64, 1)
void realnvp_flow_kernel(const float* __restrict__ x, const float* __restrict__ masks,
                         const float* __restrict__ sb1, const float* __restrict__ sb2,
                         const float* __restrict__ sb3, const float* __restrict__ scale,
                         const float* __restrict__ tb1, const float* __restrict__ tb2,
                         const float* __restrict__ tb3,
                         const __bf16* __restrict__ wp,
                         float* __restrict__ out) {
  __shared__ __attribute__((aligned(16))) __bf16 smem[2 * 2 * 16 * STR];
  const int lane = threadIdx.x & 31;
  const int wv   = threadIdx.x >> 5;
  const int g    = lane >> 4;      // half-wave
  const int ln   = lane & 15;
  __bf16* buf0 = smem + wv * (2 * 16 * STR);
  __bf16* buf1 = buf0 + 16 * STR;
  const int row0 = (blockIdx.x * 2 + wv) * 16;

  // y tile in C/D layout: yreg[ct][j] = y[row0 + j + 8g][ln + 16*ct]
  float yreg[4][8];
#pragma unroll
  for (int ct = 0; ct < 4; ++ct)
#pragma unroll
    for (int j = 0; j < 8; ++j)
      yreg[ct][j] = x[(size_t)(row0 + j + 8 * g) * D_N + ln + 16 * ct];

  float ldacc[8];
#pragma unroll
  for (int j = 0; j < 8; ++j) ldacc[j] = 0.f;

  for (int l = 0; l < L_N; ++l) {
    const __bf16* wl = wp + (size_t)l * PL;
    float mv[4], sc[4];
#pragma unroll
    for (int ct = 0; ct < 4; ++ct) {
      mv[ct] = masks[l * D_N + ln + 16 * ct];
      sc[ct] = scale[l * D_N + ln + 16 * ct];
    }

    // xm = y*m -> buf0 (bf16 row-major)
#pragma unroll
    for (int ct = 0; ct < 4; ++ct)
#pragma unroll
      for (int j = 0; j < 8; ++j)
        buf0[(j + 8 * g) * STR + ln + 16 * ct] = (__bf16)(yreg[ct][j] * mv[ct]);
    asm volatile("s_wait_dscnt 0" ::: "memory");

    // ---------------- scale net ----------------
    mlp_stage<16, 2, false>(buf0, buf1, wl + OFF_SW1, sb1 + l * H_N, lane, g, ln);
    mlp_stage<16, 8, true >(buf1, buf0, wl + OFF_SW2, sb2 + l * H_N, lane, g, ln);
    float sres[4][8];
    mlp_stage_out<8>(buf0, wl + OFF_SW3, sb3 + l * D_N, sres, lane, g, ln);
#pragma unroll
    for (int ct = 0; ct < 4; ++ct)
#pragma unroll
      for (int j = 0; j < 8; ++j)
        sres[ct][j] = (fast_tanh(sres[ct][j]) + yreg[ct][j] * mv[ct]) * sc[ct];

    // xm again -> buf0 (buf0 was consumed as h2)
#pragma unroll
    for (int ct = 0; ct < 4; ++ct)
#pragma unroll
      for (int j = 0; j < 8; ++j)
        buf0[(j + 8 * g) * STR + ln + 16 * ct] = (__bf16)(yreg[ct][j] * mv[ct]);
    asm volatile("s_wait_dscnt 0" ::: "memory");

    // ---------------- translation net ----------------
    mlp_stage<16, 2, false>(buf0, buf1, wl + OFF_TW1, tb1 + l * H_N, lane, g, ln);
    mlp_stage<16, 8, true >(buf1, buf0, wl + OFF_TW2, tb2 + l * H_N, lane, g, ln);
    float tres[4][8];
    mlp_stage_out<8>(buf0, wl + OFF_TW3, tb3 + l * D_N, tres, lane, g, ln);

    // coupling update + logdet
#pragma unroll
    for (int ct = 0; ct < 4; ++ct)
#pragma unroll
      for (int j = 0; j < 8; ++j) {
        float m = mv[ct];
        float s = sres[ct][j];
        float t = tres[ct][j] + yreg[ct][j] * m;
        float y = yreg[ct][j];
        yreg[ct][j] = m * y + (1.f - m) * (y * __expf(s) + t);
        ldacc[j] += (1.f - m) * s;
      }

    if (l + 1 < L_N) __builtin_prefetch(wl + PL, 0, 0);  // next-layer weights
  }

  // write y
#pragma unroll
  for (int ct = 0; ct < 4; ++ct)
#pragma unroll
    for (int j = 0; j < 8; ++j)
      out[(size_t)(row0 + j + 8 * g) * D_N + ln + 16 * ct] = yreg[ct][j];

  // logdet: reduce each row across its 16-lane half, then lanes 0/16 write
#pragma unroll
  for (int j = 0; j < 8; ++j) {
    float v = ldacc[j];
    v += __shfl_xor(v, 1, 32);
    v += __shfl_xor(v, 2, 32);
    v += __shfl_xor(v, 4, 32);
    v += __shfl_xor(v, 8, 32);
    if (ln == 0) out[(size_t)B_N * D_N + row0 + j + 8 * g] = v;
  }
}

// ---------------- launch ----------------
extern "C" void kernel_launch(void* const* d_in, const int* in_sizes, int n_in,
                              void* d_out, int out_size, void* d_ws, size_t ws_size,
                              hipStream_t stream) {
  const float* x     = (const float*)d_in[0];
  const float* masks = (const float*)d_in[1];
  const float* sW1   = (const float*)d_in[2];
  const float* sb1   = (const float*)d_in[3];
  const float* sW2   = (const float*)d_in[4];
  const float* sb2   = (const float*)d_in[5];
  const float* sW3   = (const float*)d_in[6];
  const float* sb3   = (const float*)d_in[7];
  const float* scale = (const float*)d_in[8];
  const float* tW1   = (const float*)d_in[9];
  const float* tb1   = (const float*)d_in[10];
  const float* tW2   = (const float*)d_in[11];
  const float* tb2   = (const float*)d_in[12];
  const float* tW3   = (const float*)d_in[13];
  const float* tb3   = (const float*)d_in[14];

  __bf16* wp = (__bf16*)d_ws;   // needs 8*196608*2 = 3.0 MiB

  const int packTot = L_N * PL;                       // 1,572,864
  pack_weights_kernel<<<(packTot + 255) / 256, 256, 0, stream>>>(
      sW1, sW2, sW3, tW1, tW2, tW3, wp);

  const int blocks = B_N / 32;                        // 2 waves * 16 rows each
  realnvp_flow_kernel<<<blocks, 64, 0, stream>>>(
      x, masks, sb1, sb2, sb3, scale, tb1, tb2, tb3, wp, (float*)d_out);
}
